// LinkPredictionGNN_49031346651174
// MI455X (gfx1250) — matrix operand order repaired and checked by reading, not compiled, not run on hardware
//
#include <hip/hip_runtime.h>

// ---- CDNA5 (gfx1250, wave32) GNN pipeline -------------------------------
// SAGEConv -> GraphNorm -> ReLU (x2) + residual linear.
// Edge aggregation: f32 gather + global_atomic_add_f32 (memory bound).
// GEMMs: v_wmma_f32_16x16x32_f16, f16 inputs / f32 accumulate.

typedef __attribute__((ext_vector_type(16))) _Float16 v16h;
typedef __attribute__((ext_vector_type(8)))  _Float16 v8h;
typedef __attribute__((ext_vector_type(8)))  float    v8f;
typedef __attribute__((ext_vector_type(4)))  float    v4f;

#define DCH 128
#define EPSV 1e-5f

// ---------------- utility kernels ----------------------------------------

__global__ __launch_bounds__(256) void gnn_fill_f32(float* __restrict__ p, float v, long long n) {
  long long i = (long long)blockIdx.x * blockDim.x + threadIdx.x;
  long long stride = (long long)gridDim.x * blockDim.x;
  for (; i < n; i += stride) p[i] = v;
}

__global__ __launch_bounds__(256) void gnn_f32_to_f16(const float* __restrict__ in,
                                                      _Float16* __restrict__ out, long long n) {
  long long i4 = ((long long)blockIdx.x * blockDim.x + threadIdx.x) * 4;
  if (i4 + 3 < n) {
    v4f v = *(const v4f*)(in + i4);
    out[i4 + 0] = (_Float16)v.x;
    out[i4 + 1] = (_Float16)v.y;
    out[i4 + 2] = (_Float16)v.z;
    out[i4 + 3] = (_Float16)v.w;
  } else {
    for (; i4 < n; ++i4) out[i4] = (_Float16)in[i4];
  }
}

// ---------------- graph aggregation --------------------------------------

__global__ __launch_bounds__(256) void gnn_degree(const int* __restrict__ ei,
                                                  float* __restrict__ cnt, int E) {
  int e = blockIdx.x * blockDim.x + threadIdx.x;
  if (e < E) atomicAdd(cnt + ei[E + e], 1.0f);
}

// one wave per edge; lane handles 4 channels (float4 gather + 4 f32 atomics)
__global__ __launch_bounds__(256) void gnn_scatter_add(const float* __restrict__ x,
                                                       const int* __restrict__ ei,
                                                       float* __restrict__ s, int E) {
  int wave = (int)(((long long)blockIdx.x * blockDim.x + threadIdx.x) >> 5);
  int lane = threadIdx.x & 31;
  if (wave >= E) return;
  int src = ei[wave];
  int dst = ei[E + wave];
  v4f xv = *(const v4f*)(x + (size_t)src * DCH + lane * 4);
  float* sp = s + (size_t)dst * DCH + lane * 4;
  atomicAdd(sp + 0, xv.x);
  atomicAdd(sp + 1, xv.y);
  atomicAdd(sp + 2, xv.z);
  atomicAdd(sp + 3, xv.w);
}

__global__ __launch_bounds__(256) void gnn_div_cvt(const float* __restrict__ s,
                                                   const float* __restrict__ cnt,
                                                   _Float16* __restrict__ aggh, long long total) {
  long long i = (long long)blockIdx.x * blockDim.x + threadIdx.x;
  if (i >= total) return;
  int n = (int)(i >> 7);
  float c = cnt[n];
  if (c < 1.0f) c = 1.0f;
  aggh[i] = (_Float16)(s[i] / c);
}

// ---------------- WMMA GEMM ----------------------------------------------
// out[row, col] = sum_k A0[row,k]*W0[col,k] (+ A1[row,k]*W1[col,k]) + bias[col] (+ extra)
// Block: 256 threads = 8 waves; block owns 16 rows, wave w owns cols 16w..16w+15.
// Per wave: 4 K-chunks of 32 per matrix pair -> v_wmma_f32_16x16x32_f16.
// Fragment layouts follow CDNA5 ISA 7.12.2 (wave32, 16-bit A 16x32 / B 32x16).

__global__ __launch_bounds__(256) void gnn_gemm_wmma(
    const _Float16* __restrict__ A0, const _Float16* __restrict__ W0,
    const _Float16* __restrict__ A1, const _Float16* __restrict__ W1,
    const float* __restrict__ bias, const float* __restrict__ extra,
    float* __restrict__ out, int N) {
  const int wv   = threadIdx.x >> 5;   // wave in block -> column tile
  const int lane = threadIdx.x & 31;
  const int half = lane >> 4;
  const int mn   = lane & 15;
  const int row0 = blockIdx.x * 16;
  const int colBase = wv * 16;
  // clamp A row so tail blocks keep EXEC all-ones around WMMA (N=50000 is exact anyway)
  const int arow = (row0 + mn < N) ? (row0 + mn) : (N - 1);

  v8f acc = {};
  {
    const _Float16* ar = A0 + (size_t)arow * DCH;
    const _Float16* wr = W0 + (size_t)(colBase + mn) * DCH;
#pragma unroll
    for (int kc = 0; kc < 4; ++kc) {
      union { v16h v; v8h h[2]; } a, b;
      // A 16x32 f16 frag: lane(m=mn), a[i]: k = kc*32 + 8*half + i (i<8), +8 more for i>=8
      a.h[0] = *(const v8h*)(ar + kc * 32 + half * 8);
      a.h[1] = *(const v8h*)(ar + kc * 32 + 16 + half * 8);
      // B 32x16 f16 frag (B[k][n] = W[col][k]): b[i]: k = kc*32 + 8*half + i (i<8), +8 more for i>=8, n = mn
      b.h[0] = *(const v8h*)(wr + kc * 32 + half * 8);
      b.h[1] = *(const v8h*)(wr + kc * 32 + 16 + half * 8);
      acc = __builtin_amdgcn_wmma_f32_16x16x32_f16(false, a.v, false, b.v,
                                                   (short)0, acc, false, false);
    }
  }
  if (A1 != nullptr) {
    const _Float16* ar = A1 + (size_t)arow * DCH;
    const _Float16* wr = W1 + (size_t)(colBase + mn) * DCH;
#pragma unroll
    for (int kc = 0; kc < 4; ++kc) {
      union { v16h v; v8h h[2]; } a, b;
      a.h[0] = *(const v8h*)(ar + kc * 32 + half * 8);
      a.h[1] = *(const v8h*)(ar + kc * 32 + 16 + half * 8);
      b.h[0] = *(const v8h*)(wr + kc * 32 + half * 8);
      b.h[1] = *(const v8h*)(wr + kc * 32 + 16 + half * 8);
      acc = __builtin_amdgcn_wmma_f32_16x16x32_f16(false, a.v, false, b.v,
                                                   (short)0, acc, false, false);
    }
  }

  // C/D layout: VGPR r holds row M = r + 8*half, col N = mn
  const int col = colBase + mn;
  const float bv = bias ? bias[col] : 0.0f;
#pragma unroll
  for (int r = 0; r < 8; ++r) {
    int row = row0 + r + half * 8;
    if (row < N) {
      size_t o = (size_t)row * DCH + col;
      float v = acc[r] + bv;
      if (extra) v += extra[o];
      out[o] = v;
    }
  }
}

// ---------------- GraphNorm ----------------------------------------------

// 128 threads/block, thread t owns column t; grid-stride over rows; one atomic per column/block
__global__ __launch_bounds__(128) void gnn_col_stats(const float* __restrict__ x,
                                                     float* __restrict__ sums, int N) {
  int c = threadIdx.x;
  float s = 0.0f, q = 0.0f;
  for (int r = blockIdx.x; r < N; r += gridDim.x) {
    float v = x[(size_t)r * DCH + c];
    s += v;
    q += v * v;
  }
  atomicAdd(&sums[c], s);
  atomicAdd(&sums[DCH + c], q);
}

// var = E[(x - a*m)^2] = E[x^2] - 2a m^2 + a^2 m^2 ; out = scale*x + shift, then relu later
__global__ __launch_bounds__(128) void gnn_norm_coeff(const float* __restrict__ sums,
                                                      const float* __restrict__ w,
                                                      const float* __restrict__ b,
                                                      const float* __restrict__ a,
                                                      float* __restrict__ coef, int N) {
  int c = threadIdx.x;
  float inv = 1.0f / (float)N;
  float mean = sums[c] * inv;
  float ex2  = sums[DCH + c] * inv;
  float al   = a[c];
  float var  = ex2 - 2.0f * al * mean * mean + al * al * mean * mean;
  float s    = w[c] * rsqrtf(var + EPSV);
  coef[c]        = s;
  coef[DCH + c]  = b[c] - s * al * mean;
}

__global__ __launch_bounds__(256) void gnn_norm_relu(const float* __restrict__ pre,
                                                     const float* __restrict__ coef,
                                                     float* __restrict__ outf,
                                                     _Float16* __restrict__ outh,
                                                     long long total) {
  long long i = (long long)blockIdx.x * blockDim.x + threadIdx.x;
  if (i >= total) return;
  int c = (int)(i & (DCH - 1));
  float v = coef[c] * pre[i] + coef[DCH + c];
  v = v > 0.0f ? v : 0.0f;
  outf[i] = v;
  if (outh) outh[i] = (_Float16)v;
}

// ---------------- host driver --------------------------------------------

extern "C" void kernel_launch(void* const* d_in, const int* in_sizes, int n_in,
                              void* d_out, int out_size, void* d_ws, size_t ws_size,
                              hipStream_t stream) {
  const float* x    = (const float*)d_in[0];
  const int*   ei   = (const int*)d_in[1];   // edge_index [2, E]: row0=src, row1=dst
  const float* Wl1  = (const float*)d_in[2];
  const float* bl1  = (const float*)d_in[3];
  const float* Wr1  = (const float*)d_in[4];
  const float* Wl2  = (const float*)d_in[5];
  const float* bl2  = (const float*)d_in[6];
  const float* Wr2  = (const float*)d_in[7];
  const float* g1w  = (const float*)d_in[8];
  const float* g1b  = (const float*)d_in[9];
  const float* g1a  = (const float*)d_in[10];
  const float* g2w  = (const float*)d_in[11];
  const float* g2b  = (const float*)d_in[12];
  const float* g2a  = (const float*)d_in[13];
  const float* Wres = (const float*)d_in[14];
  const float* bres = (const float*)d_in[15];
  float* out = (float*)d_out;

  const int N = in_sizes[0] / DCH;
  const int E = in_sizes[1] / 2;
  const long long ND = (long long)N * DCH;
  const long long WN = (long long)DCH * DCH;

  // ---- carve workspace ----
  char* ws = (char*)d_ws;
  size_t off = 0;
  auto carve = [&](size_t bytes) -> void* {
    void* p = ws + off;
    off += (bytes + 255) & ~(size_t)255;
    return p;
  };
  _Float16* xh    = (_Float16*)carve(ND * 2);
  _Float16* h1h   = (_Float16*)carve(ND * 2);
  _Float16* aggh  = (_Float16*)carve(ND * 2);
  _Float16* wl1h  = (_Float16*)carve(WN * 2);
  _Float16* wr1h  = (_Float16*)carve(WN * 2);
  _Float16* wl2h  = (_Float16*)carve(WN * 2);
  _Float16* wr2h  = (_Float16*)carve(WN * 2);
  _Float16* wresh = (_Float16*)carve(WN * 2);
  float* sbuf  = (float*)carve(ND * 4);   // scatter accumulator, later h2
  float* h1    = (float*)carve(ND * 4);
  float* hpre  = (float*)carve(ND * 4);
  float* cnt   = (float*)carve((size_t)N * 4);
  float* stats = (float*)carve(2 * DCH * 4);
  float* coef  = (float*)carve(2 * DCH * 4);

  const int ELEM_BLK = (int)((ND + 255) / 256);
  const int CVT_BLK  = (int)((ND / 4 + 255) / 256);
  const int WCVT_BLK = (int)((WN / 4 + 255) / 256);
  const int ROWT_BLK = (N + 15) / 16;
  const int EDGE_BLK = (E + 7) / 8;     // 8 waves/block, 1 wave/edge
  const int E_BLK    = (E + 255) / 256;

  // ---- precompute: f16 conversions, degrees ----
  gnn_f32_to_f16<<<CVT_BLK, 256, 0, stream>>>(x, xh, ND);
  gnn_f32_to_f16<<<WCVT_BLK, 256, 0, stream>>>(Wl1, wl1h, WN);
  gnn_f32_to_f16<<<WCVT_BLK, 256, 0, stream>>>(Wr1, wr1h, WN);
  gnn_f32_to_f16<<<WCVT_BLK, 256, 0, stream>>>(Wl2, wl2h, WN);
  gnn_f32_to_f16<<<WCVT_BLK, 256, 0, stream>>>(Wr2, wr2h, WN);
  gnn_f32_to_f16<<<WCVT_BLK, 256, 0, stream>>>(Wres, wresh, WN);
  gnn_fill_f32<<<256, 256, 0, stream>>>(cnt, 0.0f, N);
  gnn_degree<<<E_BLK, 256, 0, stream>>>(ei, cnt, E);

  // ---- layer 1: conv -> norm -> relu ----
  gnn_fill_f32<<<2048, 256, 0, stream>>>(sbuf, 0.0f, ND);
  gnn_scatter_add<<<EDGE_BLK, 256, 0, stream>>>(x, ei, sbuf, E);
  gnn_div_cvt<<<ELEM_BLK, 256, 0, stream>>>(sbuf, cnt, aggh, ND);
  gnn_gemm_wmma<<<ROWT_BLK, 256, 0, stream>>>(aggh, wl1h, xh, wr1h, bl1, nullptr, hpre, N);
  gnn_fill_f32<<<1, 256, 0, stream>>>(stats, 0.0f, 2 * DCH);
  gnn_col_stats<<<512, 128, 0, stream>>>(hpre, stats, N);
  gnn_norm_coeff<<<1, 128, 0, stream>>>(stats, g1w, g1b, g1a, coef, N);
  gnn_norm_relu<<<ELEM_BLK, 256, 0, stream>>>(hpre, coef, h1, h1h, ND);

  // ---- layer 2: conv -> norm -> relu ----
  gnn_fill_f32<<<2048, 256, 0, stream>>>(sbuf, 0.0f, ND);
  gnn_scatter_add<<<EDGE_BLK, 256, 0, stream>>>(h1, ei, sbuf, E);
  gnn_div_cvt<<<ELEM_BLK, 256, 0, stream>>>(sbuf, cnt, aggh, ND);
  gnn_gemm_wmma<<<ROWT_BLK, 256, 0, stream>>>(aggh, wl2h, h1h, wr2h, bl2, nullptr, hpre, N);
  gnn_fill_f32<<<1, 256, 0, stream>>>(stats, 0.0f, 2 * DCH);
  gnn_col_stats<<<512, 128, 0, stream>>>(hpre, stats, N);
  gnn_norm_coeff<<<1, 128, 0, stream>>>(stats, g2w, g2b, g2a, coef, N);
  gnn_norm_relu<<<ELEM_BLK, 256, 0, stream>>>(hpre, coef, sbuf, (_Float16*)nullptr, ND); // h2 -> sbuf

  // ---- residual: out = x @ Wres^T + bres + h2 ----
  gnn_gemm_wmma<<<ROWT_BLK, 256, 0, stream>>>(xh, wresh, (const _Float16*)nullptr,
                                              (const _Float16*)nullptr, bres, sbuf, out, N);
}
